// GaussianMoELayer_86174223827571
// MI455X (gfx1250) — matrix operand order repaired and checked by reading, not compiled, not run on hardware
//
#include <hip/hip_runtime.h>
#include <hip/hip_bf16.h>
#include <math.h>

#define T_TOK 4096
#define DIM   1024
#define HID   4096
#define NEXP  8
#define KTOP  2
#define XS    1032   // LDS row stride for x tile (bf16 elems), padded
#define HSRD  4104   // LDS row stride for hidden tile (bf16 elems), padded
#define LOG2PI_F 1.8378770664093453f

typedef __attribute__((ext_vector_type(16))) __bf16 v16bf;
typedef __attribute__((ext_vector_type(8)))  __bf16 v8bf;
typedef __attribute__((ext_vector_type(8)))  float  v8f;

union BFPack { v16bf v; v8bf h[2]; };

// A-operand (16-bit 16x32, ISA 7.12.2 table): lane m=l&15,
// lanes 0-15 hold K = 0..7 and 16..23, lanes 16-31 hold K = 8..15 and 24..31.
// -> two 16B chunks at +koff and +koff+16, koff = (lane<16) ? 0 : 8.
__device__ __forceinline__ v16bf load_a16(const __bf16* p, int koff) {
  BFPack u;
  u.h[0] = *(const v8bf*)(p + koff);
  u.h[1] = *(const v8bf*)(p + koff + 16);
  return u.v;
}

// B-operand (16-bit 32x16, same K order as A): column n=l&15,
// lanes 0-15 hold K = 0..7 and 16..23, lanes 16-31 hold K = 8..15 and 24..31.
// -> two 16B chunks at +kbase and +kbase+16, kbase = (lane<16) ? 0 : 8.
__device__ __forceinline__ v16bf load_b16(const __bf16* p, int kbase) {
  BFPack u;
  u.h[0] = *(const v8bf*)(p + kbase);
  u.h[1] = *(const v8bf*)(p + kbase + 16);
  return u.v;
}

// ---------------- small utility kernels ----------------

__global__ void k_zero_cnts(int* cnts) {
  if (threadIdx.x < NEXP) cnts[threadIdx.x] = 0;
}

__global__ void k_cvt_x(const float* __restrict__ x, __bf16* __restrict__ xbf, int n) {
  int i = blockIdx.x * 256 + threadIdx.x;
  if (i < n) xbf[i] = (__bf16)x[i];
}

// Transpose + f32->bf16 per expert: src [E][R][C] f32 -> dst [E][C][R] bf16
__global__ void k_transpose_bf16(const float* __restrict__ src, __bf16* __restrict__ dst,
                                 int R, int C) {
  __shared__ float tile[32][33];
  int tilesC = C >> 5;
  int perE   = (R >> 5) * tilesC;
  int bid = blockIdx.x;
  int e   = bid / perE;
  int rem = bid % perE;
  int r0  = (rem / tilesC) << 5;
  int c0  = (rem % tilesC) << 5;
  const float* s = src + (size_t)e * R * C;
  __bf16*      d = dst + (size_t)e * R * C;
  int tx = threadIdx.x, ty = threadIdx.y;          // 32 x 8
  #pragma unroll
  for (int i = 0; i < 4; ++i)
    tile[ty + 8*i][tx] = s[(size_t)(r0 + ty + 8*i) * C + c0 + tx];
  __syncthreads();
  #pragma unroll
  for (int i = 0; i < 4; ++i)
    d[(size_t)(c0 + ty + 8*i) * R + r0 + tx] = (__bf16)tile[tx][ty + 8*i];
}

// ---------------- gate: gaussian log-probs, top-2, softmax, routing ----------------

__global__ void k_gauss_route(const float* __restrict__ x, const float* __restrict__ mus,
                              const float* __restrict__ lsig,
                              float* __restrict__ logp_out, float* __restrict__ w_out,
                              float* __restrict__ idx_out,
                              int* __restrict__ cnts, int* __restrict__ bucket,
                              float* __restrict__ wgtbuf) {
  __shared__ float red[256];
  __shared__ float qs[NEXP];
  int t = blockIdx.x, tid = threadIdx.x;
  float xv[4];
  #pragma unroll
  for (int j = 0; j < 4; ++j) xv[j] = x[(size_t)t * DIM + tid + j * 256];

  for (int e = 0; e < NEXP; ++e) {
    float acc = 0.f;
    #pragma unroll
    for (int j = 0; j < 4; ++j) {
      int dd = tid + j * 256;
      float mu = mus[e * DIM + dd];
      float ls = lsig[e * DIM + dd];
      float iv = expf(-2.f * ls);
      float df = xv[j] - mu;
      acc += 0.5f * iv * df * df + ls;   // 0.5*quad + log_sigma contribution
    }
    red[tid] = acc;
    __syncthreads();
    for (int s = 128; s > 0; s >>= 1) {
      if (tid < s) red[tid] += red[tid + s];
      __syncthreads();
    }
    if (tid == 0) qs[e] = red[0];
    __syncthreads();
  }

  if (tid == 0) {
    float lp[NEXP];
    for (int e = 0; e < NEXP; ++e) {
      lp[e] = -qs[e] - 0.5f * (float)DIM * LOG2PI_F;
      logp_out[(size_t)t * NEXP + e] = lp[e];
    }
    // stable top-2 (descending, earliest index on ties)
    int i0 = 0;
    for (int e = 1; e < NEXP; ++e) if (lp[e] > lp[i0]) i0 = e;
    int i1 = (i0 == 0) ? 1 : 0;
    for (int e = 0; e < NEXP; ++e) if (e != i0 && lp[e] > lp[i1]) i1 = e;
    float e1 = expf(lp[i1] - lp[i0]);
    float w0 = 1.f / (1.f + e1);
    float w1 = e1 / (1.f + e1);
    w_out[t * KTOP + 0] = w0;  w_out[t * KTOP + 1] = w1;
    idx_out[t * KTOP + 0] = (float)i0;  idx_out[t * KTOP + 1] = (float)i1;
    int p0 = atomicAdd(&cnts[i0], 1);
    bucket[i0 * T_TOK + p0] = (t << 1) | 0;  wgtbuf[i0 * T_TOK + p0] = w0;
    int p1 = atomicAdd(&cnts[i1], 1);
    bucket[i1 * T_TOK + p1] = (t << 1) | 1;  wgtbuf[i1 * T_TOK + p1] = w1;
  }
}

// ---------------- routed expert MLP via bf16 WMMA ----------------
// One workgroup per (expert, 16-token tile). 8 waves (wave32).
// Phase B: hidden[16][HID] = gelu(X @ W1 + b1) kept entirely in LDS (bf16, ~128 KB;
//          only possible on CDNA5's 320 KB WGP LDS).
// Phase C: partial[(t*2+k)][:] = w * (hidden @ W2 + b2).

__global__ __launch_bounds__(256) void k_moe_gemm(
    const __bf16* __restrict__ xbf, const __bf16* __restrict__ W1t,
    const float* __restrict__ b1,  const __bf16* __restrict__ W2t,
    const float* __restrict__ b2,  const int* __restrict__ cnts,
    const int* __restrict__ bucket, const float* __restrict__ wgtbuf,
    float* __restrict__ partial) {
  extern __shared__ __align__(16) char smem_raw[];
  __bf16* xs  = (__bf16*)smem_raw;            // 16 x XS
  __bf16* hs  = xs + 16 * XS;                 // 16 x HSRD
  int*    stok = (int*)(hs + 16 * HSRD);      // 16 packed (t<<1)|k
  float*  swgt = (float*)(stok + 16);         // 16 combine weights

  int e    = blockIdx.x >> 8;
  int tile = blockIdx.x & 255;
  int cnt  = cnts[e];
  int t0   = tile << 4;
  if (t0 >= cnt) return;
  int nvalid = min(16, cnt - t0);

  if (threadIdx.x < 16) {
    int m = threadIdx.x;
    int src = t0 + ((m < nvalid) ? m : 0);
    stok[m] = bucket[e * T_TOK + src];
    swgt[m] = (m < nvalid) ? wgtbuf[e * T_TOK + src] : 0.f;
  }
  __syncthreads();

  // stage 16 token rows of bf16 x into LDS (uint4 = 8 bf16 per load)
  for (int idx = threadIdx.x; idx < 16 * 128; idx += 256) {
    int m = idx >> 7, c = idx & 127;
    int trow = stok[m] >> 1;
    ((uint4*)(xs + m * XS))[c] = ((const uint4*)(xbf + (size_t)trow * DIM))[c];
  }
  __syncthreads();

  int lane  = threadIdx.x & 31;
  int wv    = threadIdx.x >> 5;
  int lm    = lane & 15;
  int koffA  = (lane < 16) ? 0 : 8;    // A-operand chunk offset
  int kbaseB = (lane < 16) ? 0 : 16;   // unused contiguous K base (B loads use koffA)
  int rbase  = (lane < 16) ? 0 : 8;    // C/D row offset

  // ---- GEMM1: [16,DIM] x [DIM,HID] + b1, exact-erf GELU -> hs ----
  for (int i = 0; i < 32; ++i) {
    int n = (wv * 32 + i) * 16 + lm;
    const __bf16* bp = W1t + ((size_t)e * HID + n) * DIM;
    const __bf16* ap = xs + lm * XS;
    v8f acc = {};
    for (int k0 = 0; k0 < DIM; k0 += 32) {
      __builtin_prefetch(bp + k0 + 128, 0, 3);
      v16bf a = load_a16(ap + k0, koffA);
      v16bf b = load_b16(bp + k0, koffA);
      acc = __builtin_amdgcn_wmma_f32_16x16x32_bf16(false, a, false, b,
                                                    (short)0, acc, false, false);
    }
    float b1v = b1[e * HID + n];
    #pragma unroll
    for (int r = 0; r < 8; ++r) {
      float hv = acc[r] + b1v;
      hv = 0.5f * hv * (1.f + erff(hv * 0.70710678118654752f));
      hs[(r + rbase) * HSRD + n] = (__bf16)hv;
    }
  }
  __syncthreads();

  // ---- GEMM2: [16,HID] x [HID,DIM] + b2, scale by gate weight, scatter ----
  for (int i = 0; i < 8; ++i) {
    int n = (wv * 8 + i) * 16 + lm;
    const __bf16* bp = W2t + ((size_t)e * DIM + n) * HID;
    const __bf16* ap = hs + lm * HSRD;
    v8f acc = {};
    for (int k0 = 0; k0 < HID; k0 += 32) {
      __builtin_prefetch(bp + k0 + 128, 0, 3);
      v16bf a = load_a16(ap + k0, koffA);
      v16bf b = load_b16(bp + k0, koffA);
      acc = __builtin_amdgcn_wmma_f32_16x16x32_bf16(false, a, false, b,
                                                    (short)0, acc, false, false);
    }
    float b2v = b2[e * DIM + n];
    #pragma unroll
    for (int r = 0; r < 8; ++r) {
      int m = r + rbase;
      if (m < nvalid)
        partial[(size_t)stok[m] * DIM + n] = swgt[m] * (acc[r] + b2v);
    }
  }
}

__global__ void k_combine(const float* __restrict__ partial, float* __restrict__ y, int n) {
  int i = blockIdx.x * 256 + threadIdx.x;
  if (i < n) {
    int t = i >> 10;          // / DIM
    int d = i & (DIM - 1);
    const float* p = partial + ((size_t)t * 2) * DIM + d;
    y[i] = p[0] + p[DIM];
  }
}

// ---------------- launcher ----------------

extern "C" void kernel_launch(void* const* d_in, const int* in_sizes, int n_in,
                              void* d_out, int out_size, void* d_ws, size_t ws_size,
                              hipStream_t stream) {
  const float* x    = (const float*)d_in[0];
  const float* mus  = (const float*)d_in[1];
  const float* lsig = (const float*)d_in[2];
  const float* W1   = (const float*)d_in[3];
  const float* b1   = (const float*)d_in[4];
  const float* W2   = (const float*)d_in[5];
  const float* b2   = (const float*)d_in[6];

  char* ws = (char*)d_ws;
  __bf16* xbf    = (__bf16*)(ws);                              //   8 MB
  __bf16* W1t    = (__bf16*)(ws + (size_t)8  * 1024 * 1024);   //  64 MB [E][H][D]
  __bf16* W2t    = (__bf16*)(ws + (size_t)72 * 1024 * 1024);   //  64 MB [E][D][H]
  float*  partial= (float*) (ws + (size_t)136 * 1024 * 1024);  //  32 MB [T][2][D]
  int*    cnts   = (int*)   (ws + (size_t)168 * 1024 * 1024);
  int*    bucket = (int*)   (ws + (size_t)168 * 1024 * 1024 + 1024);
  float*  wgtbuf = (float*) (ws + (size_t)168 * 1024 * 1024 + 1024 + NEXP * T_TOK * 4);

  float* y      = (float*)d_out;                     // [T, D]
  float* logp   = y + (size_t)T_TOK * DIM;           // [T, E]
  float* wout   = logp + (size_t)T_TOK * NEXP;       // [T, 2]
  float* idxout = wout + (size_t)T_TOK * KTOP;       // [T, 2] (as float)

  k_zero_cnts<<<1, 32, 0, stream>>>(cnts);
  k_cvt_x<<<(T_TOK * DIM + 255) / 256, 256, 0, stream>>>(x, xbf, T_TOK * DIM);

  dim3 tb(32, 8);
  k_transpose_bf16<<<NEXP * (DIM / 32) * (HID / 32), tb, 0, stream>>>(W1, W1t, DIM, HID);
  k_transpose_bf16<<<NEXP * (HID / 32) * (DIM / 32), tb, 0, stream>>>(W2, W2t, HID, DIM);

  k_gauss_route<<<T_TOK, 256, 0, stream>>>(x, mus, lsig, logp, wout, idxout,
                                           cnts, bucket, wgtbuf);

  size_t smem = (size_t)16 * XS * 2 + (size_t)16 * HSRD * 2 + 16 * 4 + 16 * 4;
  k_moe_gemm<<<NEXP * 256, 256, smem, stream>>>(xbf, W1t, b1, W2t, b2,
                                                cnts, bucket, wgtbuf, partial);

  k_combine<<<(T_TOK * DIM + 255) / 256, 256, 0, stream>>>(partial, y, T_TOK * DIM);
}
